// NovaIDEIA_76373108458007
// MI455X (gfx1250) — matrix operand order repaired and checked by reading, not compiled, not run on hardware
//
#include <hip/hip_runtime.h>
#include <hip/hip_bf16.h>

typedef __attribute__((ext_vector_type(16))) _Float16 v16h;
typedef __attribute__((ext_vector_type(8)))  float    v8f;

#define NBATCH 8
#define VQ_C 16
#define VQ_M 32

// ---------------------------------------------------------------------------
// Implicit-GEMM 3D conv (and gather-form transposed conv) via WMMA f16.
// One wave computes a 16(M=C_out,padded) x 16(N=output voxels) f32 tile,
// accumulating over K = C_in*k^3 (padded to 32) with v_wmma_f32_16x16x32_f16.
//
// Per-block LDS staging (done once, outside the K loop):
//   sA   : A fragments pre-packed f16, per-lane contiguous k = 16*(lane/16) + s
//          -> inner loop reads A with ds_load_b128 x2
//   sPk  : packed (ci,kd,kh,kw) per K index (kills div/mod in the hot loop)
//   sOff : relative input offset per K index (non-transposed path)
//
// B loads are branch-free: offsets computed + clamped, 16 unconditional
// global_load_b32 issued back-to-back, then masked on convert.
// ---------------------------------------------------------------------------
template<int CIN, int COUT, int KS, int STRIDE, int PAD, bool TRANS, bool RELU>
__global__ __launch_bounds__(256)
void conv3d_wmma(const float* __restrict__ in, const float* __restrict__ w,
                 const float* __restrict__ bias, float* __restrict__ out,
                 int ID, int IH, int IW, int OD, int OH, int OW)
{
    constexpr int KTOT = CIN * KS * KS * KS;
    constexpr int KP   = (KTOT + 31) & ~31;
    constexpr int NCH  = KP / 32;

    __shared__ __align__(32) _Float16 sA[NCH * 32 * 16];
    __shared__ __align__(16) int      sPk[KP];
    __shared__ __align__(16) int      sOff[KP];

    const int inHW = IH * IW;
    const int inC  = ID * inHW;

    // ---- stage K-decode tables ----
    for (int kk = threadIdx.x; kk < KP; kk += blockDim.x) {
        int pk = -1, ro = 0;
        if (kk < KTOT) {
            int kw = kk % KS; int r = kk / KS;
            int kh = r % KS;  r /= KS;
            int kd = r % KS;  int ci = r / KS;
            pk = (ci << 24) | (kd << 16) | (kh << 8) | kw;
            ro = ci * inC + kd * inHW + kh * IW + kw;   // !TRANS relative offset
        }
        sPk[kk] = pk;
        sOff[kk] = ro;
    }
    // ---- stage pre-packed WMMA A fragments (weights, f16) ----
    for (int idx = threadIdx.x; idx < NCH * 32; idx += blockDim.x) {
        const int lane = idx & 31, ch = idx >> 5;
        const int m = lane & 15;
#pragma unroll
        for (int h = 0; h < 16; ++h) {
            const int kk = ch * 32 + ((lane < 16 ? 0 : 16) + h);
            float v = 0.0f;
            if (kk < KTOT && m < COUT) {
                int kw = kk % KS; int r = kk / KS;
                int kh = r % KS;  r /= KS;
                int kd = r % KS;  int ci = r / KS;
                v = TRANS ? w[((ci * COUT + m) * KS + kd) * (KS * KS) + kh * KS + kw]
                          : w[((m * CIN + ci) * KS + kd) * (KS * KS) + kh * KS + kw];
            }
            sA[idx * 16 + h] = (_Float16)v;
        }
    }
    __syncthreads();

    const int  lane = threadIdx.x & 31;
    const long tile = (long)blockIdx.x * (blockDim.x >> 5) + (threadIdx.x >> 5);
    const long totalPos = (long)NBATCH * OD * OH * OW;
    const long p0 = tile * 16;
    if (p0 >= totalPos) return;           // wave-uniform: EXEC stays all-1 for WMMA

    __builtin_prefetch(in, 0, 0);         // global_prefetch_b8

    long p = p0 + (lane & 15);
    const bool pvalid = (p < totalPos);
    if (!pvalid) p = totalPos - 1;
    const int ow_ = (int)(p % OW); long t = p / OW;
    const int oh_ = (int)(t % OH); t /= OH;
    const int od_ = (int)(t % OD);
    const int nb_ = (int)(t / OD);

    const float* __restrict__ inN = in + (long)nb_ * CIN * inC;

    // position terms (precomputed once)
    int odS, ohS, owS, posOff = 0;
    if constexpr (!TRANS) {
        odS = od_ * STRIDE - PAD;
        ohS = oh_ * STRIDE - PAD;
        owS = ow_ * STRIDE - PAD;
        posOff = odS * inHW + ohS * IW + owS;     // + sOff[kk] = full offset
    } else {
        odS = od_ + PAD; ohS = oh_ + PAD; owS = ow_ + PAD;
    }

    v8f acc = {};
    const int kkColBase = (lane < 16) ? 0 : 16;

#pragma unroll 1
    for (int ch = 0; ch < NCH; ++ch) {
        // A fragment: 32 bytes per lane from LDS (2x ds_load_b128)
        const v16h a = *(const v16h*)&sA[(ch * 32 + lane) * 16];

        const int kb = ch * 32 + kkColBase;
        int pk[16];
        *(int4*)&pk[0]  = *(const int4*)&sPk[kb];
        *(int4*)&pk[4]  = *(const int4*)&sPk[kb + 4];
        *(int4*)&pk[8]  = *(const int4*)&sPk[kb + 8];
        *(int4*)&pk[12] = *(const int4*)&sPk[kb + 12];

        int ro[16];
        if constexpr (!TRANS) {
            *(int4*)&ro[0]  = *(const int4*)&sOff[kb];
            *(int4*)&ro[4]  = *(const int4*)&sOff[kb + 4];
            *(int4*)&ro[8]  = *(const int4*)&sOff[kb + 8];
            *(int4*)&ro[12] = *(const int4*)&sOff[kb + 12];
        }

        // branch-free offset + validity computation
        int offs[16];
        unsigned msk = 0;
#pragma unroll
        for (int h = 0; h < 16; ++h) {
            const int c  = pk[h];
            const int ci = (c >> 24) & 255, kd = (c >> 16) & 255;
            const int kh = (c >> 8) & 255,  kw = c & 255;
            bool ok = (c >= 0) & pvalid;
            int off;
            if constexpr (!TRANS) {
                const int id_ = odS + kd, ih_ = ohS + kh, iw_ = owS + kw;
                ok = ok & ((unsigned)id_ < (unsigned)ID)
                        & ((unsigned)ih_ < (unsigned)IH)
                        & ((unsigned)iw_ < (unsigned)IW);
                off = posOff + ro[h];
            } else {
                const int td = odS - kd, th = ohS - kh, tw = owS - kw;
                ok = ok && (td >= 0) && (th >= 0) && (tw >= 0)
                        && (td % STRIDE == 0) && (th % STRIDE == 0) && (tw % STRIDE == 0);
                const int id_ = td / STRIDE, ih_ = th / STRIDE, iw_ = tw / STRIDE;
                ok = ok && (id_ < ID) && (ih_ < IH) && (iw_ < IW);
                off = ci * inC + id_ * inHW + ih_ * IW + iw_;
            }
            offs[h] = ok ? off : 0;       // clamp: always a safe address
            msk |= (ok ? 1u : 0u) << h;
        }

        // batched unconditional loads (single wait, many in flight)
        float bv[16];
#pragma unroll
        for (int h = 0; h < 16; ++h) bv[h] = inN[offs[h]];

        v16h bm;
#pragma unroll
        for (int h = 0; h < 16; ++h)
            bm[h] = (_Float16)(((msk >> h) & 1u) ? bv[h] : 0.0f);

        acc = __builtin_amdgcn_wmma_f32_16x16x32_f16(
            /*neg_a=*/false, a, /*neg_b=*/false, bm,
            /*c_mod=*/(short)0, acc, /*reuse_a=*/false, /*reuse_b=*/false);
    }

    // ---- epilogue: bias (+relu), scatter C tile (32-bit offsets) ----
    const int outHW = OH * OW;
    const int outC  = OD * outHW;
    float* __restrict__ outN = out + (long)nb_ * COUT * outC;
    const int spOff = od_ * outHW + oh_ * OW + ow_;
#pragma unroll
    for (int r = 0; r < 8; ++r) {
        const int m = r + ((lane < 16) ? 0 : 8);
        if (pvalid && m < COUT) {
            float v = acc[r] + bias[m];
            if (RELU) v = fmaxf(v, 0.0f);
            outN[m * outC + spOff] = v;
        }
    }
}

// ---------------------------------------------------------------------------
// VQ: nearest code per voxel-vector (C=16, M=32 codes).
// Codebook is staged to LDS with the CDNA5 async-to-LDS path
// (GLOBAL_LOAD_ASYNC_TO_LDS_B32, tracked by ASYNCcnt; ISA ch.10 / 08_async).
// ---------------------------------------------------------------------------
__global__ __launch_bounds__(256)
void vq_kernel(const float* __restrict__ z, const float* __restrict__ emb,
               float* __restrict__ quant, int* __restrict__ codes,
               float* __restrict__ counts, float* __restrict__ sse,
               int D, int H, int W)
{
    __shared__ __align__(16) float se[VQ_M * VQ_C];   // 512 floats = 2KB

    // Async copy: 2 rounds of 256 lanes x b32. LDS VDST = addr[31:0] of the
    // generic pointer (flat->LDS truncation rule, ISA 10.2).
    {
        const int tid = threadIdx.x;
        unsigned           lds0 = (unsigned)(unsigned long long)&se[tid];
        unsigned long long g0   = (unsigned long long)&emb[tid];
        asm volatile("global_load_async_to_lds_b32 %0, %1, off"
                     :: "v"(lds0), "v"(g0) : "memory");
        unsigned           lds1 = (unsigned)(unsigned long long)&se[tid + 256];
        unsigned long long g1   = (unsigned long long)&emb[tid + 256];
        asm volatile("global_load_async_to_lds_b32 %0, %1, off"
                     :: "v"(lds1), "v"(g1) : "memory");
        asm volatile("s_wait_asynccnt 0x0" ::: "memory");
    }
    __syncthreads();

    const long total = (long)NBATCH * D * H * W;
    const long p = (long)blockIdx.x * blockDim.x + threadIdx.x;
    float err = 0.0f;
    if (p < total) {
        int w_ = (int)(p % W); long t = p / W;
        int h_ = (int)(t % H); t /= H;
        int d_ = (int)(t % D); int n_ = (int)(t / D);
        const int  sp   = D * H * W;
        const long base = (long)n_ * VQ_C * sp + d_ * H * W + h_ * W + w_;

        float xv[VQ_C];
#pragma unroll
        for (int c = 0; c < VQ_C; ++c) xv[c] = z[base + (long)c * sp];

        int best = 0; float bestd = 3.4e38f;
#pragma unroll 4
        for (int mc = 0; mc < VQ_M; ++mc) {
            float d2 = 0.0f;
#pragma unroll
            for (int c = 0; c < VQ_C; ++c) {
                float diff = xv[c] - se[mc * VQ_C + c];
                d2 = fmaf(diff, diff, d2);
            }
            if (d2 < bestd) { bestd = d2; best = mc; }
        }
#pragma unroll
        for (int c = 0; c < VQ_C; ++c) {
            float e = se[best * VQ_C + c];
            quant[base + (long)c * sp] = e;
            float diff = e - xv[c];
            err = fmaf(diff, diff, err);
        }
        codes[p] = best;
        atomicAdd(&counts[best], 1.0f);
    }
    for (int off = 16; off > 0; off >>= 1) err += __shfl_xor(err, off);
    if ((threadIdx.x & 31) == 0) atomicAdd(sse, err);
}

// ---------------------------------------------------------------------------
__global__ void vq_finalize(const float* __restrict__ counts,
                            const float* __restrict__ sse,
                            const float* __restrict__ cluster_size,
                            float* __restrict__ outv,
                            long off_loss, long off_perp, long off_used, float ntot)
{
    const int i = threadIdx.x;  // 0..31
    float pr   = counts[i] / ntot;
    float term = pr * logf(pr + 1e-10f);
    for (int off = 16; off > 0; off >>= 1) term += __shfl_xor(term, off);
    float used = (cluster_size[i] > 1e-5f) ? 1.0f : 0.0f;
    for (int off = 16; off > 0; off >>= 1) used += __shfl_xor(used, off);
    if (i == 0) {
        outv[off_loss] = sse[0] / (ntot * (float)VQ_C);
        outv[off_perp] = expf(-term);
        outv[off_used] = used / (float)VQ_M;
    }
}

__global__ void zero_small(float* ws) { ws[threadIdx.x] = 0.0f; }

// ---------------------------------------------------------------------------
extern "C" void kernel_launch(void* const* d_in, const int* in_sizes, int n_in,
                              void* d_out, int out_size, void* d_ws, size_t ws_size,
                              hipStream_t stream)
{
    (void)in_sizes; (void)n_in; (void)out_size; (void)ws_size;

    const float* x   = (const float*)d_in[0];
    // d_in[1] = epoch (unused: reference takes the full VQ branch regardless)
    const float* w1  = (const float*)d_in[2];  const float* b1  = (const float*)d_in[3];
    const float* w2  = (const float*)d_in[4];  const float* b2  = (const float*)d_in[5];
    const float* w3  = (const float*)d_in[6];  const float* b3  = (const float*)d_in[7];
    const float* emb = (const float*)d_in[8];  const float* csz = (const float*)d_in[9];
    const float* wd1 = (const float*)d_in[10]; const float* bd1 = (const float*)d_in[11];
    const float* wd2 = (const float*)d_in[12]; const float* bd2 = (const float*)d_in[13];
    const float* wd3 = (const float*)d_in[14]; const float* bd3 = (const float*)d_in[15];

    // workspace layout (floats)
    float* ws     = (float*)d_ws;
    float* counts = ws;            // 32
    float* sse    = ws + 32;       // 1 (padded)
    float* z1 = ws + 64;           // (8,4,32,64,64)   = 4,194,304
    float* z2 = z1 + 4194304L;     // (8,8,16,32,32)   = 1,048,576
    float* z3 = z2 + 1048576L;     // (8,16,16,32,32)  = 2,097,152
    float* qz = z3 + 2097152L;     // quantized, same  = 2,097,152
    float* y1 = qz + 2097152L;     // (8,8,32,64,64)   = 8,388,608
    float* y2 = y1 + 8388608L;     // (8,4,64,128,128) = 33,554,432

    float* outp = (float*)d_out;
    const long OUTE   = 25165824L;   // (8,3,64,128,128)
    const long NCODES = 131072L;     // (8,16,32,32)
    int* codes = (int*)(outp + OUTE + 1);

    auto blocksFor = [](long totalPos) {
        long tiles = (totalPos + 15) / 16;   // 16 output voxels per wave
        return (int)((tiles + 7) / 8);       // 8 waves per 256-thread block
    };

    zero_small<<<1, 64, 0, stream>>>(ws);

    // --- encoder ---
    conv3d_wmma<3, 4, 4, 2, 1, false, true><<<blocksFor(8L*32*64*64), 256, 0, stream>>>(
        x,  w1, b1, z1, 64, 128, 128, 32, 64, 64);
    conv3d_wmma<4, 8, 4, 2, 1, false, true><<<blocksFor(8L*16*32*32), 256, 0, stream>>>(
        z1, w2, b2, z2, 32, 64, 64, 16, 32, 32);
    conv3d_wmma<8, 16, 3, 1, 1, false, false><<<blocksFor(8L*16*32*32), 256, 0, stream>>>(
        z2, w3, b3, z3, 16, 32, 32, 16, 32, 32);

    // --- VQ ---
    vq_kernel<<<(int)((NCODES + 255) / 256), 256, 0, stream>>>(
        z3, emb, qz, codes, counts, sse, 16, 32, 32);

    // --- decoder (transposed convs in gather form) ---
    conv3d_wmma<16, 8, 4, 2, 1, true, true><<<blocksFor(8L*32*64*64), 256, 0, stream>>>(
        qz, wd1, bd1, y1, 16, 32, 32, 32, 64, 64);
    conv3d_wmma<8, 4, 4, 2, 1, true, true><<<blocksFor(8L*64*128*128), 256, 0, stream>>>(
        y1, wd2, bd2, y2, 32, 64, 64, 64, 128, 128);
    conv3d_wmma<4, 3, 3, 1, 1, true, false><<<blocksFor(8L*64*128*128), 256, 0, stream>>>(
        y2, wd3, bd3, outp, 64, 128, 128, 64, 128, 128);

    // --- scalars ---
    vq_finalize<<<1, 32, 0, stream>>>(counts, sse, csz, outp,
        OUTE, OUTE + 1 + NCODES, OUTE + 1 + NCODES + 1, (float)NCODES);
}
